// LinearZeRO3_27041114096128
// MI455X (gfx1250) — matrix operand order repaired and checked by reading, not compiled, not run on hardware
//
#include <hip/hip_runtime.h>
#include <hip/hip_bf16.h>

typedef __attribute__((ext_vector_type(16))) _Float16 v16h;
typedef __attribute__((ext_vector_type(8)))  _Float16 v8h;
typedef __attribute__((ext_vector_type(8)))  float    v8f;
typedef __attribute__((ext_vector_type(4)))  float    f4;

union Frag16 { v16h v; v8h h[2]; };

#define BM 256             // workgroup tile rows (M)
#define BN 128             // workgroup tile cols (N)
#define BK 32              // K chunk (f16 WMMA K)
#define LDSS 40            // halves per LDS row: 32 data + 8 pad (80B, 16B aligned)
#define NTHREADS 256       // 8 waves: 4 along M (64 rows each) x 2 along N (64 cols each)

// pack 8 f32 -> 8 f16
__device__ __forceinline__ v8h cvt8(f4 a, f4 b) {
  v8h r;
  r[0] = (_Float16)a.x; r[1] = (_Float16)a.y; r[2] = (_Float16)a.z; r[3] = (_Float16)a.w;
  r[4] = (_Float16)b.x; r[5] = (_Float16)b.y; r[6] = (_Float16)b.z; r[7] = (_Float16)b.w;
  return r;
}

// 32 consecutive f32 -> 4 x v8h
__device__ __forceinline__ void stage_a(const float* __restrict__ g, v8h st[4]) {
  const f4* p = (const f4*)g;
  f4 x0 = p[0], x1 = p[1], x2 = p[2], x3 = p[3];
  f4 x4 = p[4], x5 = p[5], x6 = p[6], x7 = p[7];
  st[0] = cvt8(x0, x1); st[1] = cvt8(x2, x3);
  st[2] = cvt8(x4, x5); st[3] = cvt8(x6, x7);
}

// 16 consecutive f32 -> 2 x v8h
__device__ __forceinline__ void stage_b(const float* __restrict__ g, v8h st[2]) {
  const f4* p = (const f4*)g;
  f4 x0 = p[0], x1 = p[1], x2 = p[2], x3 = p[3];
  st[0] = cvt8(x0, x1); st[1] = cvt8(x2, x3);
}

// One K-chunk step with compile-time double-buffer parity: no runtime buffer
// select math in the hot loop (kills the v_mad/s_xor + WMMA-hazard v_nops).
template <int CUR>
__device__ __forceinline__ void k_step(
    int kt, int NK,
    const float* __restrict__ aG, const float* __restrict__ bG,
    _Float16 (*As)[BM * LDSS], _Float16 (*Bs)[BN * LDSS],
    int aSRow, int bSRow, int bSCol,
    int aRow0, int aKb, int bCol0, int bKb,
    v8f (&acc)[4][4]) {
  constexpr int NXT = CUR ^ 1;
  const bool hasNext = (kt + 1) < NK;

  // issue next chunk's global loads early; converts overlap via staggered waits
  v8h aStage[4], bStage[2];
  if (hasNext) {
    stage_a(aG + (kt + 1) * BK, aStage);
    stage_b(bG + (kt + 1) * BK, bStage);
  }

  // gather fragments from current buffer (16B-aligned ds_load_b128)
  Frag16 aF[4], bF[4];
#pragma unroll
  for (int fm = 0; fm < 4; ++fm) {
    const _Float16* p = &As[CUR][(aRow0 + fm * 16) * LDSS + aKb];
    aF[fm].h[0] = *(const v8h*)(p);        // K = kb .. kb+7
    aF[fm].h[1] = *(const v8h*)(p + 16);   // K = kb+16 .. kb+23
  }
#pragma unroll
  for (int fn = 0; fn < 4; ++fn) {
    const _Float16* p = &Bs[CUR][(bCol0 + fn * 16) * LDSS + bKb];
    bF[fn].h[0] = *(const v8h*)(p);        // K = kb .. kb+7
    bF[fn].h[1] = *(const v8h*)(p + 16);   // K = kb+16 .. kb+23
  }

  // 16 WMMAs: 64x64 wave tile
#pragma unroll
  for (int fm = 0; fm < 4; ++fm)
#pragma unroll
    for (int fn = 0; fn < 4; ++fn)
      acc[fm][fn] = __builtin_amdgcn_wmma_f32_16x16x32_f16(
          false, aF[fm].v, false, bF[fn].v,
          (short)0, acc[fm][fn], false, false);

  if (hasNext) {
    // write next chunk into the other buffer (overlaps with WMMA drain)
    _Float16* as = &As[NXT][aSRow * LDSS];
    *(v8h*)(as)      = aStage[0];
    *(v8h*)(as + 8)  = aStage[1];
    *(v8h*)(as + 16) = aStage[2];
    *(v8h*)(as + 24) = aStage[3];
    _Float16* bs = &Bs[NXT][bSRow * LDSS + bSCol];
    *(v8h*)(bs)     = bStage[0];
    *(v8h*)(bs + 8) = bStage[1];
    __syncthreads();   // next buffer written + current buffer reads complete
  }
}

__global__ __launch_bounds__(NTHREADS, 2)
void linear_zero3_wmma(const float* __restrict__ X,   // [M, K] row-major
                       const float* __restrict__ W,   // [N, K] row-major
                       float* __restrict__ Y,         // [M, N] row-major
                       int M, int N, int K) {
  // double-buffered tiles: 2*(256+128)*40 halves = 60 KB
  __shared__ __attribute__((aligned(16))) _Float16 As[2][BM * LDSS];
  __shared__ __attribute__((aligned(16))) _Float16 Bs[2][BN * LDSS];

  const int tid  = threadIdx.x;
  const int lane = tid & 31;
  const int wave = tid >> 5;
  const int wm   = wave & 3;       // 4 slabs of 64 rows
  const int wn   = wave >> 2;      // 2 slabs of 64 cols

  const int mBase = blockIdx.y * BM;
  const int nBase = blockIdx.x * BN;

  // cooperative staging: A -> 1 thread per row (32 f32); B -> 2 threads per row (16 f32)
  const int aSRow = tid;                // 0..255
  const int bSRow = tid >> 1;           // 0..127
  const int bSCol = (tid & 1) << 4;     // 0 or 16

  const float* aG = X + (size_t)(mBase + aSRow) * K;
  const float* bG = W + (size_t)(nBase + bSRow) * K + bSCol;

  // per-lane fragment gather bases (ISA wave32 VGPR layouts)
  const int l15   = lane & 15;
  const int aRow0 = wm * 64 + l15;            // + fm*16
  const int aKb   = (lane & 16) ? 8 : 0;      // A: lanes 16-31 start at K=8
  const int bCol0 = wn * 64 + l15;            // + fn*16
  const int bKb   = (lane & 16) ? 8 : 0;      // B: lanes 16-31 start at K=8

  v8f acc[4][4] = {};

  // prologue: stage K-chunk 0 into buffer 0
  {
    v8h aStage[4], bStage[2];
    stage_a(aG, aStage);
    stage_b(bG, bStage);
    _Float16* as = &As[0][aSRow * LDSS];
    *(v8h*)(as)      = aStage[0];
    *(v8h*)(as + 8)  = aStage[1];
    *(v8h*)(as + 16) = aStage[2];
    *(v8h*)(as + 24) = aStage[3];
    _Float16* bs = &Bs[0][bSRow * LDSS + bSCol];
    *(v8h*)(bs)     = bStage[0];
    *(v8h*)(bs + 8) = bStage[1];
  }
  __syncthreads();

  const int NK = K / BK;   // 128 (even)
  for (int kt = 0; kt < NK; kt += 2) {
    k_step<0>(kt,     NK, aG, bG, As, Bs, aSRow, bSRow, bSCol,
              aRow0, aKb, bCol0, bKb, acc);
    k_step<1>(kt + 1, NK, aG, bG, As, Bs, aSRow, bSRow, bSCol,
              aRow0, aKb, bCol0, bKb, acc);
  }

  // epilogue: C/D layout — lane holds col = l15, VGPR r holds row r (+8 upper half-wave)
  const unsigned rShift = (lane >> 4) << 3;
  const unsigned uN = (unsigned)N;
#pragma unroll
  for (int fm = 0; fm < 4; ++fm) {
#pragma unroll
    for (int fn = 0; fn < 4; ++fn) {
      const unsigned col = (unsigned)(nBase + wn * 64 + fn * 16 + l15);
      const unsigned row = (unsigned)(mBase + wm * 64 + fm * 16) + rShift;
      unsigned idx = row * uN + col;          // fits in 32 bits (M*N = 2^25)
#pragma unroll
      for (int r = 0; r < 8; ++r) {
        __builtin_nontemporal_store(acc[fm][fn][r], Y + idx);
        idx += uN;
      }
    }
  }
}

extern "C" void kernel_launch(void* const* d_in, const int* in_sizes, int n_in,
                              void* d_out, int out_size, void* d_ws, size_t ws_size,
                              hipStream_t stream) {
  (void)n_in; (void)out_size; (void)d_ws; (void)ws_size;
  const float* X = (const float*)d_in[0];   // [B,S,D_IN] = [M,K]
  const float* W = (const float*)d_in[1];   // [D_OUT,D_IN] = [N,K]
  float* Y = (float*)d_out;                 // [M,N]

  const int K = 4096;
  const int N = in_sizes[1] / K;            // 4096
  const int M = in_sizes[0] / K;            // 8192

  dim3 grid(N / BN, M / BM);                // (32, 32)
  linear_zero3_wmma<<<grid, NTHREADS, 0, stream>>>(X, W, Y, M, N, K);
}
